// LLaDARouter_24936580120992
// MI455X (gfx1250) — matrix operand order repaired and checked by reading, not compiled, not run on hardware
//
#include <hip/hip_runtime.h>
#include <hip/hip_bf16.h>
#include <math.h>

// ---------------- types for CDNA5 WMMA ----------------
typedef __attribute__((ext_vector_type(16))) __bf16          v16bf;
typedef __attribute__((ext_vector_type(8)))  float           v8f;
typedef __attribute__((ext_vector_type(16))) unsigned short  v16u16;
typedef int v4i __attribute__((vector_size(16)));            // matches builtin param type

union BF16x16 {
    v16u16       u;
    v16bf        b;
    uint4        q[2];
    unsigned int d[8];
};

// Problem constants
#define D_IN   2048
#define H_DIM  256
#define N_EXP  16
#define T_TOK  16384          // 4 * 4096
#define KBLKS  64             // D_IN / 32
#define TM_WG  128            // tokens per workgroup (8 waves x 16)

// Feature detection (guarded: falls back to known-good paths if absent)
#if __has_builtin(__builtin_amdgcn_global_load_async_to_lds_b128) && \
    __has_builtin(__builtin_amdgcn_s_wait_asynccnt)
#define USE_ASYNC_LDS 1
#else
#define USE_ASYNC_LDS 0
#endif

#if USE_ASYNC_LDS
// async 16B copy: global (AS1) -> LDS (AS3), tracked by ASYNCcnt
__device__ __forceinline__ void async_cp16(const unsigned short* gsrc, unsigned short* ldst) {
    __builtin_amdgcn_global_load_async_to_lds_b128(
        (__attribute__((address_space(1))) v4i*)(gsrc),
        (__attribute__((address_space(3))) v4i*)(ldst), 0, 0);
}
#endif

// fp32 -> bf16 round-to-nearest-even (scalar fallback)
__device__ __forceinline__ unsigned short f2b(float f) {
    unsigned int u = __float_as_uint(f);
    u += 0x7fffu + ((u >> 16) & 1u);
    return (unsigned short)(u >> 16);
}

// packed fp32x2 -> bf16x2 (one dword, low = a, high = b)
#if __has_builtin(__builtin_amdgcn_cvt_pk_bf16_f32)
__device__ __forceinline__ unsigned int pk2b(float a, float b) {
    auto p = __builtin_amdgcn_cvt_pk_bf16_f32(a, b);   // v_cvt_pk_bf16_f32
    return __builtin_bit_cast(unsigned int, p);
}
#else
__device__ __forceinline__ unsigned int pk2b(float a, float b) {
    return (unsigned int)f2b(a) | ((unsigned int)f2b(b) << 16);
}
#endif

// 16-lane (half-wave) reductions: xor masks 1,2,4,8 stay within each 16-lane half
__device__ __forceinline__ float redsum16(float v) {
    v += __shfl_xor(v, 1, 32);
    v += __shfl_xor(v, 2, 32);
    v += __shfl_xor(v, 4, 32);
    v += __shfl_xor(v, 8, 32);
    return v;
}
__device__ __forceinline__ float redmax16(float v) {
    v = fmaxf(v, __shfl_xor(v, 1, 32));
    v = fmaxf(v, __shfl_xor(v, 2, 32));
    v = fmaxf(v, __shfl_xor(v, 4, 32));
    v = fmaxf(v, __shfl_xor(v, 8, 32));
    return v;
}

// -------- kernel 1: convert Wp (K-blocked) + Wg to bf16, zero accumulators --------
// Wpb layout: [kb][h][k32]  (kb = 0..63, h = 0..255, k = 0..31)  -> coalesced 16KB block per k-step
__global__ void llada_cvt_kernel(const float* __restrict__ Wp,
                                 const float* __restrict__ Wg,
                                 unsigned short* __restrict__ Wpb,
                                 unsigned short* __restrict__ Wgb,
                                 float* __restrict__ accum) {
    int idx = blockIdx.x * 256 + threadIdx.x;
    if (idx < 17) accum[idx] = 0.0f;                   // [0]=z-loss sum, [1..16]=per-expert load
    if (idx < H_DIM * D_IN) {                          // 524288
        int kb  = idx >> 13;                           // / (256*32)
        int rem = idx & 8191;
        int h   = rem >> 5;
        int k   = rem & 31;
        Wpb[idx] = f2b(Wp[h * D_IN + kb * 32 + k]);
    } else {
        int j = idx - H_DIM * D_IN;
        if (j < N_EXP * H_DIM) Wgb[j] = f2b(Wg[j]);    // Wgb[e][h], row-major
    }
}

// -------- kernel 2: fused projection GEMM + LayerNorm + router GEMM + softmax/top2 --------
__global__ __launch_bounds__(256)
void llada_router_main(const float* __restrict__ x,
                       const float* __restrict__ ln_w,
                       const float* __restrict__ ln_b,
                       const float* __restrict__ temperature,
                       const unsigned short* __restrict__ Wpb,
                       const unsigned short* __restrict__ Wgb,
                       float* __restrict__ out_rw,
                       float* __restrict__ out_disp,
                       float* __restrict__ accum) {
    // 64KB LDS, time-multiplexed:
    //   phase 1: first 16384 ushorts = double-buffered Wp bf16 block (2 x [h=256][k=32])
    //   phase 2: 32768 ushorts = normalized activations bf16 [128 tokens][256 h]
    __shared__ unsigned short smem[32768];

    const int tid     = threadIdx.x;
    const int wave    = tid >> 5;
    const int lane    = tid & 31;
    const int col     = lane & 15;               // N index for B/C fragments, M index for A
    const int halfOff = (lane < 16) ? 0 : 8;     // C-tile row offset per ISA layout
    const int kA      = (lane < 16) ? 0 : 16;    // A-frag K chunk base (contiguous 16-k per half)
    const int kB      = (lane < 16) ? 0 : 16;    // B-frag K chunk base
    const int wg      = blockIdx.x;

    // ---------------- phase 1: proj = x @ Wp^T via bf16 WMMA ----------------
    v8f acc[16];
#pragma unroll
    for (int nt = 0; nt < 16; ++nt)
#pragma unroll
        for (int i = 0; i < 8; ++i) acc[nt][i] = 0.0f;

    const int tokRow = wg * TM_WG + wave * 16 + col;       // A-fragment row (m = col)
    const float* xrow = x + (size_t)tokRow * D_IN;

#if USE_ASYNC_LDS
    // prologue: async-copy k-block 0 into buffer 0 (64B per thread, ASYNCcnt-tracked)
    {
        const unsigned short* gsrc = Wpb + tid * 32;
        unsigned short* ldst = smem + tid * 32;
#pragma unroll
        for (int j = 0; j < 4; ++j)
            async_cp16(gsrc + j * 8, ldst + j * 8);
    }
#endif

    for (int kb = 0; kb < KBLKS; ++kb) {
#if USE_ASYNC_LDS
        __builtin_amdgcn_s_wait_asynccnt(0);   // own async loads for buf[kb&1] complete
        __syncthreads();                       // everyone's complete; buf[(kb+1)&1] free
        if (kb + 1 < KBLKS) {                  // overlap next weight tile with compute
            const unsigned short* gsrc = Wpb + (kb + 1) * 8192 + tid * 32;
            unsigned short* ldst = smem + ((kb + 1) & 1) * 8192 + tid * 32;
#pragma unroll
            for (int j = 0; j < 4; ++j)
                async_cp16(gsrc + j * 8, ldst + j * 8);
        }
        const unsigned short* sB = smem + (kb & 1) * 8192;
#else
        __syncthreads();  // previous block's LDS reads complete
        {   // stage 16KB of Wp-bf16: thread tid copies 64 contiguous bytes (coalesced)
            const uint4* src = (const uint4*)(Wpb + kb * 8192 + tid * 32);
            uint4* dst = (uint4*)(smem + tid * 32);
            uint4 t0 = src[0], t1 = src[1], t2 = src[2], t3 = src[3];
            dst[0] = t0; dst[1] = t1; dst[2] = t2; dst[3] = t3;
        }
        if (kb + 1 < KBLKS)
            __builtin_prefetch(Wpb + (kb + 1) * 8192 + tid * 32, 0, 3);  // global_prefetch
        __syncthreads();
        const unsigned short* sB = smem;
#endif

        // A fragment: x fp32 -> bf16 in registers, contiguous k = 16*half + s (same order as B)
        BF16x16 A;
        const float* ap = xrow + kb * 32 + kA;
        float4 f0 = *(const float4*)(ap + 0);
        float4 f1 = *(const float4*)(ap + 4);
        float4 f2 = *(const float4*)(ap + 8);
        float4 f3 = *(const float4*)(ap + 12);
        A.d[0] = pk2b(f0.x, f0.y); A.d[1] = pk2b(f0.z, f0.w);
        A.d[2] = pk2b(f1.x, f1.y); A.d[3] = pk2b(f1.z, f1.w);
        A.d[4] = pk2b(f2.x, f2.y); A.d[5] = pk2b(f2.z, f2.w);
        A.d[6] = pk2b(f3.x, f3.y); A.d[7] = pk2b(f3.z, f3.w);

#pragma unroll
        for (int nt = 0; nt < 16; ++nt) {
            BF16x16 B;
            const uint4* bp = (const uint4*)(sB + (nt * 16 + col) * 32 + kB);
            B.q[0] = bp[0];
            B.q[1] = bp[1];
            acc[nt] = __builtin_amdgcn_wmma_f32_16x16x32_bf16(
                false, A.b, false, B.b, (short)0, acc[nt], false, false);
        }
    }

    // ---------------- LayerNorm in registers (rows live in lane halves) ----------------
    float mu[8], rs[8];
#pragma unroll
    for (int r = 0; r < 8; ++r) {
        float s = 0.0f, ss = 0.0f;
#pragma unroll
        for (int nt = 0; nt < 16; ++nt) {
            float v = acc[nt][r];
            s += v; ss += v * v;
        }
        s  = redsum16(s);
        ss = redsum16(ss);
        float m   = s * (1.0f / 256.0f);
        float var = ss * (1.0f / 256.0f) - m * m;
        mu[r] = m;
        rs[r] = rsqrtf(var + 1e-5f);
    }

    __syncthreads();   // everyone done reading the Wp stage buffers before overwriting

    // normalized -> bf16 into LDS (A-operand storage for second GEMM)
#pragma unroll
    for (int nt = 0; nt < 16; ++nt) {
        int h = nt * 16 + col;
        float lw  = ln_w[h];
        float lbv = ln_b[h];
#pragma unroll
        for (int r = 0; r < 8; ++r) {
            float nv = (acc[nt][r] - mu[r]) * rs[r] * lw + lbv;
            smem[(wave * 16 + r + halfOff) * H_DIM + h] = f2b(nv);
        }
    }
    __syncthreads();

    // ---------------- phase 2: logits = norm @ Wg^T via bf16 WMMA (K=256) ----------------
    v8f accL;
#pragma unroll
    for (int i = 0; i < 8; ++i) accL[i] = 0.0f;

    const int nbase = wave * 16 * H_DIM;
#pragma unroll
    for (int kb2 = 0; kb2 < 8; ++kb2) {
        BF16x16 A2, B2;
        int base = nbase + col * H_DIM + kb2 * 32 + kA;          // m = col
        A2.q[0] = *(const uint4*)(smem + base);
        A2.q[1] = *(const uint4*)(smem + base + 8);
        const uint4* bp2 = (const uint4*)(Wgb + col * H_DIM + kb2 * 32 + kB);
        B2.q[0] = bp2[0];
        B2.q[1] = bp2[1];
        accL = __builtin_amdgcn_wmma_f32_16x16x32_bf16(
            false, A2.b, false, B2.b, (short)0, accL, false, false);
    }

    // ---------------- phase 3: softmax, top-2, dispatch, loss partials ----------------
    const float ts = 1.0f / (fabsf(temperature[0]) + 1e-6f);
    float zpart = 0.0f, loadpart = 0.0f;
    const int tbase = wg * TM_WG + wave * 16 + halfOff;

#pragma unroll
    for (int r = 0; r < 8; ++r) {
        float l = accL[r] * ts;                 // scaled logit for (token, expert=col)
        zpart += l * l;
        float mx = redmax16(l);
        float e  = __expf(l - mx);
        float s  = redsum16(e);
        float w  = e / s;                       // routing weight
        loadpart += w;
        int t = tbase + r;
        out_rw[t * N_EXP + col] = w;

        // top-1 (lowest index wins ties, like jax top_k)
        float w1 = w; int i1 = col;
#pragma unroll
        for (int mm = 1; mm <= 8; mm <<= 1) {
            float ow = __shfl_xor(w1, mm, 32);
            int   oi = __shfl_xor(i1, mm, 32);
            if (ow > w1 || (ow == w1 && oi < i1)) { w1 = ow; i1 = oi; }
        }
        // top-2
        float wmk = (col == i1) ? -1.0f : w;
        float w2 = wmk; int i2 = col;
#pragma unroll
        for (int mm = 1; mm <= 8; mm <<= 1) {
            float ow = __shfl_xor(w2, mm, 32);
            int   oi = __shfl_xor(i2, mm, 32);
            if (ow > w2 || (ow == w2 && oi < i2)) { w2 = ow; i2 = oi; }
        }
        float rn = 1.0f / (w1 + w2 + 1e-6f);
        float dv = (col == i1) ? w1 * rn : ((col == i2) ? w2 * rn : 0.0f);
        out_disp[t * N_EXP + col] = dv;
    }

    // z-loss: full-wave reduce then one atomic
    float zw = zpart;
#pragma unroll
    for (int mm = 1; mm <= 16; mm <<= 1) zw += __shfl_xor(zw, mm, 32);
    if (lane == 0) atomicAdd(&accum[0], zw);

    // per-expert load: fold the two halves (same expert, different tokens), 16 atomics/wave
    loadpart += __shfl_xor(loadpart, 16, 32);
    if (lane < 16) atomicAdd(&accum[1 + col], loadpart);
}

// -------- kernel 3: combine loss accumulators --------
__global__ void llada_finalize(const float* __restrict__ accum, float* __restrict__ out_loss) {
    if (blockIdx.x == 0 && threadIdx.x == 0) {
        float z = accum[0] / (float)(T_TOK * N_EXP);
        const float ideal = 1.0f / (float)N_EXP;
        float lbsum = 0.0f;
        for (int e = 0; e < N_EXP; ++e) {
            float al = accum[1 + e] / (float)T_TOK;   // actual mean load of expert e
            lbsum += ideal * (logf(ideal) - logf(al));
        }
        float lb = lbsum / (float)N_EXP;
        *out_loss = 0.005f * z + 0.005f * lb;
    }
}

extern "C" void kernel_launch(void* const* d_in, const int* in_sizes, int n_in,
                              void* d_out, int out_size, void* d_ws, size_t ws_size,
                              hipStream_t stream) {
    const float* x    = (const float*)d_in[0];   // [4,4096,2048]
    const float* Wp   = (const float*)d_in[1];   // [256,2048]
    const float* Wg   = (const float*)d_in[2];   // [16,256]
    const float* ln_w = (const float*)d_in[3];   // [256]
    const float* ln_b = (const float*)d_in[4];   // [256]
    const float* temp = (const float*)d_in[5];   // [1]

    // workspace layout: bf16 Wp (1MB) | bf16 Wg (8KB) | 17 float accumulators
    unsigned short* Wpb = (unsigned short*)d_ws;
    unsigned short* Wgb = Wpb + (H_DIM * D_IN);
    float* accum = (float*)((char*)d_ws + (size_t)H_DIM * D_IN * 2 + (size_t)N_EXP * H_DIM * 2);

    float* out_rw   = (float*)d_out;                 // [T, E]
    float* out_disp = out_rw + (size_t)T_TOK * N_EXP;
    float* out_loss = out_rw + 2 * (size_t)T_TOK * N_EXP;

    const int cvtN = H_DIM * D_IN + N_EXP * H_DIM;   // 528384
    llada_cvt_kernel<<<dim3((cvtN + 255) / 256), dim3(256), 0, stream>>>(Wp, Wg, Wpb, Wgb, accum);
    llada_router_main<<<dim3(T_TOK / TM_WG), dim3(256), 0, stream>>>(
        x, ln_w, ln_b, temp, Wpb, Wgb, out_rw, out_disp, accum);
    llada_finalize<<<dim3(1), dim3(32), 0, stream>>>(accum, out_loss);
}
